// Model_28578712387584
// MI455X (gfx1250) — matrix operand order repaired and checked by reading, not compiled, not run on hardware
//
#include <hip/hip_runtime.h>
#include <hip/hip_bf16.h>
#include <math.h>

typedef __attribute__((ext_vector_type(16))) _Float16 v16h;
typedef __attribute__((ext_vector_type(8)))  float    v8f;

union Frag16 { v16h h; uint4 q[2]; };

// ---------------------------------------------------------------------------
// f32 -> f16 convert with K padding (zeros) ; 2D grid: y = row, x covers Kp
// ---------------------------------------------------------------------------
__global__ void cvt_pad_f16(const float* __restrict__ src, _Float16* __restrict__ dst,
                            int K, int Kp) {
    int c = blockIdx.x * blockDim.x + threadIdx.x;
    int r = blockIdx.y;
    if (c >= Kp) return;
    float v = (c < K) ? src[(size_t)r * K + c] : 0.0f;
    dst[(size_t)r * Kp + c] = (_Float16)v;
}

__global__ void zero_f32(float* __restrict__ p, int n) {
    int i = blockIdx.x * blockDim.x + threadIdx.x;
    if (i < n) p[i] = 0.0f;
}

// ---------------------------------------------------------------------------
// WMMA f16 GEMM:  C[M,N] = A[M,K] * W[N,K]^T + bias
// block = 256 threads (8 waves); block tile = 128(M) x 64(N); wave = 16x64.
// The 64xK weight tile is staged once into LDS (stride K+8 halves to keep
// b128 reads bank-conflict-free), shared by all 8 waves for the whole K loop.
// A is streamed from global with a one-step register pipeline + prefetch.
// K multiple of 32. A,W row-major f16 (K padded). C row-major f32 (pre-BN).
// ---------------------------------------------------------------------------
__global__ __launch_bounds__(256)
void gemm_wmma_f16(const _Float16* __restrict__ A, const _Float16* __restrict__ W,
                   const float* __restrict__ bias, float* __restrict__ C,
                   int N, int K) {
    extern __shared__ _Float16 lw[];          // [64][K+8]
    const int LDK  = K + 8;                   // pad: lanes land on distinct banks
    const int lane = threadIdx.x & 31;
    const int wave = threadIdx.x >> 5;
    const int l16  = lane & 15;
    const int half = lane >> 4;
    const int m0   = blockIdx.x * 128 + wave * 16;
    const int n0   = blockIdx.y * 64;

    // ---- stage W tile [n0..n0+63] x K into LDS (uniform trip count) ----
    {
        const int chunksPerRow = K >> 3;      // uint4 = 8 halves
        const int total = 64 * chunksPerRow;  // divisible by 256 for K=320/512
        for (int idx = threadIdx.x; idx < total; idx += 256) {
            int col = idx / chunksPerRow;
            int kk  = (idx - col * chunksPerRow) << 3;
            uint4 v = *(const uint4*)(W + (size_t)(n0 + col) * K + kk);
            *(uint4*)(&lw[col * LDK + kk]) = v;
        }
    }
    __syncthreads();

    const _Float16* arow = A + (size_t)(m0 + l16) * K;

    v8f acc[4];
#pragma unroll
    for (int t = 0; t < 4; ++t) acc[t] = (v8f){0.f,0.f,0.f,0.f,0.f,0.f,0.f,0.f};

    // A fragment pipeline: chunks at K offsets half*8 and 16+half*8
    Frag16 a_cur;
    a_cur.q[0] = *(const uint4*)(arow + half * 8);
    a_cur.q[1] = *(const uint4*)(arow + 16 + half * 8);

    for (int k0 = 0; k0 < K; k0 += 32) {
        Frag16 a_nxt;
        int kn = (k0 + 32 < K) ? (k0 + 32) : k0;     // clamp: uniform, no branch
        a_nxt.q[0] = *(const uint4*)(arow + kn + half * 8);
        a_nxt.q[1] = *(const uint4*)(arow + kn + 16 + half * 8);
        __builtin_prefetch(arow + kn + 64, 0, 1);    // global_prefetch_b8 on A stream

#pragma unroll
        for (int t = 0; t < 4; ++t) {
            // B fragment from LDS: col = t*16+l16, K chunks at half*8 and 16+half*8
            const _Float16* bp = &lw[(t * 16 + l16) * LDK + k0 + half * 8];
            Frag16 b;
            b.q[0] = *(const uint4*)(bp);
            b.q[1] = *(const uint4*)(bp + 16);
            acc[t] = __builtin_amdgcn_wmma_f32_16x16x32_f16(
                false, a_cur.h, false, b.h, (short)0, acc[t], false, false);
        }
        a_cur = a_nxt;
    }

    // Epilogue: C/D layout -> VGPR r holds row m0 + half*8 + r, col = n_tile + l16
#pragma unroll
    for (int t = 0; t < 4; ++t) {
        int col = n0 + t * 16 + l16;
        float bv = bias[col];
#pragma unroll
        for (int r = 0; r < 8; ++r) {
            int row = m0 + half * 8 + r;
            C[(size_t)row * N + col] = acc[t][r] + bv;
        }
    }
}

// ---------------------------------------------------------------------------
// Per-column sum / sum-of-squares partials (BN training-mode batch stats)
// ---------------------------------------------------------------------------
__global__ void col_stats(const float* __restrict__ Hm, float* __restrict__ sum,
                          float* __restrict__ sq, int N, int rowsPerSeg) {
    int col = blockIdx.x * blockDim.x + threadIdx.x;
    int r0  = blockIdx.y * rowsPerSeg;
    float s = 0.f, q = 0.f;
    for (int r = r0; r < r0 + rowsPerSeg; ++r) {
        float v = Hm[(size_t)r * N + col];
        s += v; q += v * v;
    }
    atomicAdd(&sum[col], s);
    atomicAdd(&sq[col],  q);
}

// ---------------------------------------------------------------------------
// BN (batch stats, biased var) + ReLU, output f16 (for next GEMM). N pow2.
// ---------------------------------------------------------------------------
__global__ void bn_relu_f16(const float* __restrict__ Hm, const float* __restrict__ sum,
                            const float* __restrict__ sq, const float* __restrict__ g,
                            const float* __restrict__ b, _Float16* __restrict__ out,
                            int N, float invB) {
    size_t i = (size_t)blockIdx.x * blockDim.x + threadIdx.x;
    int col = (int)(i & (size_t)(N - 1));
    float mu  = sum[col] * invB;
    float var = sq[col] * invB - mu * mu;
    float v = (Hm[i] - mu) * rsqrtf(var + 1e-3f) * g[col] + b[col];
    out[i] = (_Float16)fmaxf(v, 0.0f);
}

// BN + ReLU in place, f32 (final embeddings live in d_out). N pow2.
__global__ void bn_relu_inplace(float* __restrict__ Hm, const float* __restrict__ sum,
                                const float* __restrict__ sq, const float* __restrict__ g,
                                const float* __restrict__ b, int N, float invB) {
    size_t i = (size_t)blockIdx.x * blockDim.x + threadIdx.x;
    int col = (int)(i & (size_t)(N - 1));
    float mu  = sum[col] * invB;
    float var = sq[col] * invB - mu * mu;
    float v = (Hm[i] - mu) * rsqrtf(var + 1e-3f) * g[col] + b[col];
    Hm[i] = fmaxf(v, 0.0f);
}

// ---------------------------------------------------------------------------
// Per-row L2 distance: ||i_e - e + 1e-6||_2 over D=1024. One wave per row.
// ---------------------------------------------------------------------------
__global__ __launch_bounds__(256)
void row_dist(const float* __restrict__ ie, const float* __restrict__ e,
              float* __restrict__ dist, int D) {
    int lane = threadIdx.x & 31;
    int row  = blockIdx.x * 8 + (threadIdx.x >> 5);
    const float* pi = ie + (size_t)row * D;
    const float* pe = e  + (size_t)row * D;
    float s = 0.f;
    for (int c = lane; c < D; c += 32) {
        float d = pi[c] - pe[c] + 1e-6f;
        s += d * d;
    }
#pragma unroll
    for (int off = 16; off; off >>= 1) s += __shfl_down(s, off, 32);
    if (lane == 0) dist[row] = sqrtf(s);
}

// count((d_neg - d_pos) > 1.0) -> *out (float, pre-zeroed)
__global__ void margin_count(const float* __restrict__ dpos, const float* __restrict__ dneg,
                             float* __restrict__ out) {
    int i = blockIdx.x * blockDim.x + threadIdx.x;
    int lane = threadIdx.x & 31;
    float v = ((dneg[i] - dpos[i]) > 1.0f) ? 1.0f : 0.0f;
#pragma unroll
    for (int off = 16; off; off >>= 1) v += __shfl_down(v, off, 32);
    if (lane == 0 && v != 0.0f) atomicAdd(out, v);
}

// ---------------------------------------------------------------------------
extern "C" void kernel_launch(void* const* d_in, const int* in_sizes, int n_in,
                              void* d_out, int out_size, void* d_ws, size_t ws_size,
                              hipStream_t stream) {
    (void)in_sizes; (void)n_in; (void)out_size; (void)ws_size;
    const int B = 32768, T = 300, Tp = 320, Hd = 512, D = 1024;
    const float invB = 1.0f / (float)B;

    const float* i_e    = (const float*)d_in[0];
    const float* tags[2] = { (const float*)d_in[1], (const float*)d_in[2] };
    // d_in[3]=lat, d_in[4]=lon unused by the output
    const float* fc1_w = (const float*)d_in[5];
    const float* fc1_b = (const float*)d_in[6];
    const float* bn1_g = (const float*)d_in[7];
    const float* bn1_b = (const float*)d_in[8];
    const float* fc2_w = (const float*)d_in[9];
    const float* fc2_b = (const float*)d_in[10];
    const float* bn2_g = (const float*)d_in[11];
    const float* bn2_b = (const float*)d_in[12];

    float* out = (float*)d_out; // [i_e | tp_e | tn_e | correct]

    // workspace layout (256B-aligned offsets)
    char* ws = (char*)d_ws;
    _Float16* w1h   = (_Float16*)(ws);                                   // 512*320*2
    _Float16* w2h   = (_Float16*)(ws + 327680);                          // 1024*512*2
    _Float16* x16   = (_Float16*)(ws + 1376256);                         // 32768*320*2
    float*    h1    = (float*)   (ws + 22347776);                        // 32768*512*4
    _Float16* a1h   = (_Float16*)(ws + 89456640);                        // 32768*512*2
    float*    sum1  = (float*)   (ws + 123011072);                       // 512
    float*    sq1   = sum1 + 512;
    float*    sum2  = sq1  + 512;                                        // 1024
    float*    sq2   = sum2 + 1024;
    float*    dists = (float*)   (ws + 123023360);                       // 2*32768

    const size_t lds1 = (size_t)64 * (Tp + 8) * sizeof(_Float16);        // 41,984 B
    const size_t lds2 = (size_t)64 * (Hd + 8) * sizeof(_Float16);        // 66,560 B

    // weights -> f16 (K padded to mult of 32)
    cvt_pad_f16<<<dim3((Tp + 255) / 256, Hd), 256, 0, stream>>>(fc1_w, w1h, T, Tp);
    cvt_pad_f16<<<dim3((Hd + 255) / 256, D),  256, 0, stream>>>(fc2_w, w2h, Hd, Hd);

    // i_e passthrough to output slot 0
    hipMemcpyAsync(out, i_e, (size_t)B * D * sizeof(float),
                   hipMemcpyDeviceToDevice, stream);
    // zero the 'correct' accumulator
    zero_f32<<<1, 32, 0, stream>>>(out + (size_t)3 * B * D, 1);

    for (int t = 0; t < 2; ++t) {
        float* e    = out + (size_t)(1 + t) * B * D; // tp_e / tn_e slot
        float* dist = dists + (size_t)t * B;

        // input -> f16, padded K
        cvt_pad_f16<<<dim3((Tp + 255) / 256, B), 256, 0, stream>>>(tags[t], x16, T, Tp);

        // layer 1: GEMM + batch stats + BN/ReLU -> f16 activations
        zero_f32<<<(3072 + 255) / 256, 256, 0, stream>>>(sum1, 3072); // sum1..sq2
        gemm_wmma_f16<<<dim3(B / 128, Hd / 64), 256, lds1, stream>>>(x16, w1h, fc1_b, h1, Hd, Tp);
        col_stats<<<dim3(Hd / 256, 32), 256, 0, stream>>>(h1, sum1, sq1, Hd, B / 32);
        bn_relu_f16<<<(int)(((size_t)B * Hd) / 256), 256, 0, stream>>>(
            h1, sum1, sq1, bn1_g, bn1_b, a1h, Hd, invB);

        // layer 2: GEMM straight into d_out slot, then BN/ReLU in place
        gemm_wmma_f16<<<dim3(B / 128, D / 64), 256, lds2, stream>>>(a1h, w2h, fc2_b, e, D, Hd);
        col_stats<<<dim3(D / 256, 32), 256, 0, stream>>>(e, sum2, sq2, D, B / 32);
        bn_relu_inplace<<<(int)(((size_t)B * D) / 256), 256, 0, stream>>>(
            e, sum2, sq2, bn2_g, bn2_b, D, invB);

        // pairwise distance vs i_e
        row_dist<<<B / 8, 256, 0, stream>>>(i_e, e, dist, D);
    }

    margin_count<<<B / 256, 256, 0, stream>>>(dists, dists + B, out + (size_t)3 * B * D);
}
